// EEG_AMD_3Class_57260503990390
// MI455X (gfx1250) — compile-verified, matrix operand order checked by reading, not run on hardware
//
#include <hip/hip_runtime.h>
#include <hip/hip_bf16.h>

typedef __attribute__((ext_vector_type(16))) _Float16 v16h;
typedef __attribute__((ext_vector_type(8)))  float    v8f;

namespace {
constexpr int cB = 64, cN = 19, cSEG = 15, cT = 400;
constexpr int cFIN = 64, cHID = 128;
constexpr int cM = cB * cSEG * cN;   // 18240 GEMM rows (divisible by 16)
}

// ---------------- WMMA fragment helpers (CDNA5 16x16x32 f16) ----------------
// A (MxK): lane<16 holds row M=lane%16, K = {kb..kb+7, kb+16..kb+23};
// lane>=16 holds the same rows with K = {kb+8..kb+15, kb+24..kb+31}.
// B is fed "column-major": lane holds output-column N=lane%16 == weight row of a
// row-major [N,K] weight matrix, same K split -> contiguous 16B loads.
__device__ __forceinline__ v16h frag_h(const _Float16* row, int kbase, int kh) {
  const _Float16* p = row + kbase + kh * 8;
  v16h r;
#pragma unroll
  for (int i = 0; i < 8; ++i) { r[i] = p[i]; r[i + 8] = p[i + 16]; }
  return r;
}
__device__ __forceinline__ v8f wmma_f16(v16h a, v16h b, v8f c) {
  return __builtin_amdgcn_wmma_f32_16x16x32_f16(false, a, false, b, (short)0, c, false, false);
}

// ---------------- weight prep: f32 -> f16 (one-shot, tiny) ----------------
__global__ __launch_bounds__(256) void cvt_kernel(const float* __restrict__ src,
                                                  _Float16* __restrict__ dst, int n) {
  const int i = blockIdx.x * blockDim.x + threadIdx.x;
  if (i < n) dst[i] = (_Float16)src[i];
}
// conv1 weights [32][25] -> [32][32] zero-padded in K
__global__ __launch_bounds__(256) void pad_w1_kernel(const float* __restrict__ w1,
                                                     _Float16* __restrict__ w1h) {
  const int i = blockIdx.x * blockDim.x + threadIdx.x;
  if (i < 32 * 32) {
    const int c = i >> 5, kk = i & 31;
    w1h[i] = (_Float16)(kk < 25 ? w1[c * 25 + kk] : 0.f);
  }
}

// ---------------- Stage 1: temporal CNN, both convs as implicit-GEMM WMMA ----------------
// One WG (4 waves) per sequence. conv1: im2col A1[128x32] (100 valid pos, K=25 padded
// to 32; rows padded to 128 so each wave owns a uniform 2x2 tile block).
// out = A1 @ w1h^T -> bias+relu+pool4 fused on fragments (8 consecutive p per lane).
// conv2: im2col A2[16x288], out = A2 @ w2h^T -> bias+relu+pool2+mean fused.
__global__ __launch_bounds__(128) void temporal_kernel(
    const float* __restrict__ X,
    const _Float16* __restrict__ w1h, const float* __restrict__ b1,
    const _Float16* __restrict__ w2h, const float* __restrict__ b2,
    _Float16* __restrict__ feath) {
  const int seq = blockIdx.x;                 // 0 .. 18239
  const int b = seq / (cSEG * cN);
  const int r0 = seq % (cSEG * cN);
  const int s = r0 / cN, n = r0 % cN;
  const int tid = threadIdx.x;
  const int lane = tid & 31, wv = tid >> 5;
  const int idx = lane & 15, kh = lane >> 4;

  __shared__ float    xin[cT];          // 400 f32
  __shared__ _Float16 A1[128 * 32];     // conv1 im2col (f16), rows 100..127 zero
  __shared__ float    p1s[32 * 25];     // after relu+pool4
  __shared__ _Float16 A2[16 * 288];     // conv2 im2col (f16)
  __shared__ float    part[64][2];      // pool2 partial sums per channel

  const float* xptr = X + (((size_t)b * cN + n) * cSEG + s) * cT;  // X is [B,N,SEG,T]
  for (int i = tid; i < cT; i += 128) xin[i] = xptr[i];
  __syncthreads();

  // im2col conv1: A1[p][kk] = x[p*4-12+kk], zero pad (p>=100 or kk>=25 or OOB)
  for (int i = tid; i < 128 * 32; i += 128) {
    const int p = i >> 5, kk = i & 31;
    float v = 0.f;
    if (p < 100 && kk < 25) {
      const int ip = p * 4 - 12 + kk;
      if (ip >= 0 && ip < cT) v = xin[ip];
    }
    A1[i] = (_Float16)v;
  }
  __syncthreads();

  // conv1 WMMA: 8 M-tiles x 2 N-tiles; wave wv owns M-tiles {2wv, 2wv+1} x N-tiles {0,1}
  {
    const v16h bf0 = frag_h(&w1h[(0 * 16 + idx) * 32], 0, kh);
    const v16h bf1 = frag_h(&w1h[(1 * 16 + idx) * 32], 0, kh);
    const v16h af0 = frag_h(&A1[((wv * 2 + 0) * 16 + idx) * 32], 0, kh);
    const v16h af1 = frag_h(&A1[((wv * 2 + 1) * 16 + idx) * 32], 0, kh);
#pragma unroll
    for (int t = 0; t < 4; ++t) {
      const int dm = t >> 1, nt = t & 1;
      const int mt = wv * 2 + dm;
      v8f accv = {};
      accv = wmma_f16(dm ? af1 : af0, nt ? bf1 : bf0, accv);
      const int c = nt * 16 + idx;
      const float bc = b1[c];
      float v[8];
#pragma unroll
      for (int r = 0; r < 8; ++r) v[r] = fmaxf(accv[r] + bc, 0.f);
      const int base = mt * 16 + 8 * kh;      // 8 consecutive positions per lane
      const int j0 = base >> 2;               // pool4 window index
      if (j0 < 25)
        p1s[c * 25 + j0] = fmaxf(fmaxf(v[0], v[1]), fmaxf(v[2], v[3]));
      if (j0 + 1 < 25)
        p1s[c * 25 + j0 + 1] = fmaxf(fmaxf(v[4], v[5]), fmaxf(v[6], v[7]));
    }
  }
  __syncthreads();

  // im2col conv2: A2[p][ci*9+kk] = p1[ci][p*2-4+kk], zero pad
  for (int i = tid; i < 16 * 288; i += 128) {
    const int p = i / 288, col = i % 288;
    const int ci = col / 9, kk = col % 9;
    float v = 0.f;
    if (p < 13) {
      const int ip = p * 2 - 4 + kk;
      if (ip >= 0 && ip < 25) v = p1s[ci * 25 + ip];
    }
    A2[i] = (_Float16)v;
  }
  __syncthreads();

  // conv2 WMMA: 1 M-tile x 4 N-tiles (one per wave), K=288
  {
    const int nt = wv;
    v8f accv = {};
#pragma unroll
    for (int kb = 0; kb < 288; kb += 32)
      accv = wmma_f16(frag_h(&A2[idx * 288], kb, kh),
                      frag_h(&w2h[(nt * 16 + idx) * 288], kb, kh), accv);
    const int c = nt * 16 + idx;
    const float bc = b2[c];
    float v[8];
#pragma unroll
    for (int r = 0; r < 8; ++r) v[r] = fmaxf(accv[r] + bc, 0.f);
    // pool2 (VALID: windows (0,1)..(10,11); p=12 dropped) + partial sum for mean
    float sres;
    if (kh == 0)   // p = 0..7  -> windows 0..3
      sres = fmaxf(v[0], v[1]) + fmaxf(v[2], v[3]) + fmaxf(v[4], v[5]) + fmaxf(v[6], v[7]);
    else           // p = 8..12 -> windows 4,5
      sres = fmaxf(v[0], v[1]) + fmaxf(v[2], v[3]);
    part[c][kh] = sres;
  }
  __syncthreads();
  if (tid < 64) {
    const float f = (part[tid][0] + part[tid][1]) * (1.f / 6.f);
    feath[((size_t)(b * cSEG + s) * cN + n) * 64 + tid] = (_Float16)f;
  }
}

// ---------------- Stage 2/4: WMMA GEMM  Y[M,N] = A[M,K] * W[N,K]^T (all f16 in) ----------------
__global__ __launch_bounds__(128) void gemm_f16w(
    const _Float16* __restrict__ A, const _Float16* __restrict__ W, float* __restrict__ Y,
    int M, int Kt, int Nt, int tilesTotal) {
  const int lane = threadIdx.x & 31;
  const int gw = blockIdx.x * (blockDim.x >> 5) + (threadIdx.x >> 5);
  if (gw >= tilesTotal) return;                 // wave-uniform
  const int tilesN = Nt >> 4;
  const int mt = gw / tilesN, nt = gw % tilesN;
  const int idx = lane & 15, kh = lane >> 4;
  const _Float16* arow = A + (size_t)(mt * 16 + idx) * Kt;
  const _Float16* wrow = W + (size_t)(nt * 16 + idx) * Kt;
  v8f accv = {};
  for (int kb = 0; kb < Kt; kb += 32)
    accv = wmma_f16(frag_h(arow, kb, kh), frag_h(wrow, kb, kh), accv);
#pragma unroll
  for (int r = 0; r < 8; ++r)
    Y[(size_t)(mt * 16 + r + 8 * kh) * Nt + nt * 16 + idx] = accv[r];
}

// ---------------- Stage 3/5: hop-mix (Abar = sum_k alpha_k A_k) + BN + ReLU ----------------
__global__ __launch_bounds__(128) void gcn_mix_kernel(
    const float* __restrict__ Ahat, const float* __restrict__ q, const float* __restrict__ Y,
    const float* __restrict__ bng, const float* __restrict__ bnb,
    const float* __restrict__ bnm, const float* __restrict__ bnv,
    float* __restrict__ H, _Float16* __restrict__ Hh, int F) {
  const int bs = blockIdx.x;            // 0..959
  const int b = bs / cSEG, s = bs % cSEG;
  const int tid = threadIdx.x;
  __shared__ float Ab[cN * cN];
  __shared__ float Ys[cN * cHID];

  const float q0 = q[0], q1 = q[1], q2 = q[2], q3 = q[3];
  const float mx = fmaxf(fmaxf(q0, q1), fmaxf(q2, q3));
  const float e0 = __expf(q0 - mx), e1 = __expf(q1 - mx), e2 = __expf(q2 - mx), e3 = __expf(q3 - mx);
  const float inv = 1.f / (e0 + e1 + e2 + e3);
  const float al[4] = {e0 * inv, e1 * inv, e2 * inv, e3 * inv};

  for (int i = tid; i < cN * cN; i += 128) {
    float acc = 0.f;
#pragma unroll
    for (int k = 0; k < 4; ++k)
      acc += al[k] * Ahat[(((size_t)b * 4 + k) * cSEG + s) * (cN * cN) + i];
    Ab[i] = acc;
  }
  const float* yb = Y + (size_t)bs * cN * F;
  for (int i = tid; i < cN * F; i += 128) Ys[i] = yb[i];
  __syncthreads();

  for (int i = tid; i < cN * F; i += 128) {
    const int nr = i / F, o = i % F;
    float acc = 0.f;
#pragma unroll
    for (int m = 0; m < cN; ++m) acc += Ab[nr * cN + m] * Ys[m * F + o];
    float hv = (acc - bnm[o]) * rsqrtf(bnv[o] + 1e-5f) * bng[o] + bnb[o];
    hv = fmaxf(hv, 0.f);
    const size_t oi = (size_t)bs * cN * F + i;
    if (H)  H[oi] = hv;
    if (Hh) Hh[oi] = (_Float16)hv;
  }
}

// ---------------- Stage 6: node mean + cls token + positional embedding ----------------
__global__ __launch_bounds__(64) void tokens_kernel(
    const float* __restrict__ H2, const float* __restrict__ cls,
    const float* __restrict__ pos, float* __restrict__ tokens) {
  const int b = blockIdx.x, d = threadIdx.x;    // 64 threads = DM
  tokens[(size_t)(b * 16 + 0) * 64 + d] = cls[d] + pos[d];
  for (int s = 0; s < cSEG; ++s) {
    float acc = 0.f;
    const float* hp = H2 + ((size_t)(b * cSEG + s) * cN) * 64 + d;
#pragma unroll
    for (int n = 0; n < cN; ++n) acc += hp[n * 64];
    tokens[(size_t)(b * 16 + s + 1) * 64 + d] = acc * (1.f / 19.f) + pos[(s + 1) * 64 + d];
  }
}

// ---------------- Stage 7: fused transformer layer (one WG = one batch element) ----------------
__global__ __launch_bounds__(128) void tx_layer(
    float* __restrict__ tokens,
    const _Float16* __restrict__ wqkv, const float* __restrict__ bqkv,
    const _Float16* __restrict__ wo,  const float* __restrict__ bo,
    const float* __restrict__ ln1g, const float* __restrict__ ln1b,
    const _Float16* __restrict__ w1,  const float* __restrict__ b1v,
    const _Float16* __restrict__ w2,  const float* __restrict__ b2v,
    const float* __restrict__ ln2g, const float* __restrict__ ln2b) {
  const int b = blockIdx.x;
  const int tid = threadIdx.x;
  const int lane = tid & 31, wv = tid >> 5;
  const int idx = lane & 15, kh = lane >> 4;

  __shared__ float    xs[16][64];
  __shared__ _Float16 xh[16][64];
  __shared__ float    qkv_s[16][192];
  __shared__ float    sc[4][16][16];
  __shared__ _Float16 ah[16][64];
  __shared__ float    tms[16][64];
  __shared__ float    mu_s[16], rs_s[16];
  __shared__ _Float16 ffh[16][128];

  float* xrow = tokens + (size_t)b * 16 * 64;
  for (int i = tid; i < 1024; i += 128) {
    const float v = xrow[i];
    (&xs[0][0])[i] = v;
    (&xh[0][0])[i] = (_Float16)v;
  }
  __syncthreads();

  // ---- QKV: [16,64] @ [192,64]^T, 12 N-tiles, 3 per wave ----
  {
    const v16h a0 = frag_h(&xh[idx][0], 0, kh);
    const v16h a1 = frag_h(&xh[idx][0], 32, kh);
#pragma unroll
    for (int j = 0; j < 3; ++j) {
      const int nt = wv * 3 + j;
      const _Float16* wr = wqkv + (size_t)(nt * 16 + idx) * 64;
      v8f accv = {};
      accv = wmma_f16(a0, frag_h(wr, 0, kh), accv);
      accv = wmma_f16(a1, frag_h(wr, 32, kh), accv);
      const float bias = bqkv[nt * 16 + idx];
#pragma unroll
      for (int r = 0; r < 8; ++r) qkv_s[r + 8 * kh][nt * 16 + idx] = accv[r] + bias;
    }
  }
  __syncthreads();

  // ---- attention scores (dh=16, scale 1/4) ----
  for (int i = tid; i < 1024; i += 128) {
    const int h = i >> 8, r = (i >> 4) & 15, c = i & 15;
    float d = 0.f;
#pragma unroll
    for (int k = 0; k < 16; ++k) d += qkv_s[r][h * 16 + k] * qkv_s[c][64 + h * 16 + k];
    sc[h][r][c] = d * 0.25f;
  }
  __syncthreads();
  if (tid < 64) {
    const int h = tid >> 4, r = tid & 15;
    float mx = -1e30f;
#pragma unroll
    for (int c = 0; c < 16; ++c) mx = fmaxf(mx, sc[h][r][c]);
    float ssum = 0.f;
#pragma unroll
    for (int c = 0; c < 16; ++c) { const float e = __expf(sc[h][r][c] - mx); sc[h][r][c] = e; ssum += e; }
    const float inv = 1.f / ssum;
#pragma unroll
    for (int c = 0; c < 16; ++c) sc[h][r][c] *= inv;
  }
  __syncthreads();
  for (int i = tid; i < 1024; i += 128) {
    const int h = i >> 8, r = (i >> 4) & 15, d = i & 15;
    float acc = 0.f;
#pragma unroll
    for (int c = 0; c < 16; ++c) acc += sc[h][r][c] * qkv_s[c][128 + h * 16 + d];
    ah[r][h * 16 + d] = (_Float16)acc;
  }
  __syncthreads();

  // ---- out-proj + residual ----
  {
    const v16h a0 = frag_h(&ah[idx][0], 0, kh);
    const v16h a1 = frag_h(&ah[idx][0], 32, kh);
    const int nt = wv;
    const _Float16* wr = wo + (size_t)(nt * 16 + idx) * 64;
    v8f accv = {};
    accv = wmma_f16(a0, frag_h(wr, 0, kh), accv);
    accv = wmma_f16(a1, frag_h(wr, 32, kh), accv);
    const float bias = bo[nt * 16 + idx];
#pragma unroll
    for (int r = 0; r < 8; ++r) {
      const int row = r + 8 * kh;
      tms[row][nt * 16 + idx] = accv[r] + bias + xs[row][nt * 16 + idx];
    }
  }
  __syncthreads();

  // ---- LN1 -> xs/xh ----
  if (tid < 16) {
    float mu = 0.f;
#pragma unroll
    for (int d = 0; d < 64; ++d) mu += tms[tid][d];
    mu *= (1.f / 64.f);
    float var = 0.f;
#pragma unroll
    for (int d = 0; d < 64; ++d) { const float t = tms[tid][d] - mu; var += t * t; }
    mu_s[tid] = mu;
    rs_s[tid] = rsqrtf(var * (1.f / 64.f) + 1e-5f);
  }
  __syncthreads();
  for (int i = tid; i < 1024; i += 128) {
    const int r = i >> 6, d = i & 63;
    const float v = (tms[r][d] - mu_s[r]) * rs_s[r] * ln1g[d] + ln1b[d];
    xs[r][d] = v;
    xh[r][d] = (_Float16)v;
  }
  __syncthreads();

  // ---- FF1: [16,64] @ [128,64]^T, relu ----
  {
    const v16h a0 = frag_h(&xh[idx][0], 0, kh);
    const v16h a1 = frag_h(&xh[idx][0], 32, kh);
#pragma unroll
    for (int j = 0; j < 2; ++j) {
      const int nt = wv * 2 + j;
      const _Float16* wr = w1 + (size_t)(nt * 16 + idx) * 64;
      v8f accv = {};
      accv = wmma_f16(a0, frag_h(wr, 0, kh), accv);
      accv = wmma_f16(a1, frag_h(wr, 32, kh), accv);
      const float bias = b1v[nt * 16 + idx];
#pragma unroll
      for (int r = 0; r < 8; ++r)
        ffh[r + 8 * kh][nt * 16 + idx] = (_Float16)fmaxf(accv[r] + bias, 0.f);
    }
  }
  __syncthreads();

  // ---- FF2: [16,128] @ [64,128]^T + residual ----
  {
    const int nt = wv;
    const _Float16* wr = w2 + (size_t)(nt * 16 + idx) * 128;
    v8f accv = {};
#pragma unroll
    for (int kb = 0; kb < 128; kb += 32)
      accv = wmma_f16(frag_h(&ffh[idx][0], kb, kh), frag_h(wr, kb, kh), accv);
    const float bias = b2v[nt * 16 + idx];
#pragma unroll
    for (int r = 0; r < 8; ++r) {
      const int row = r + 8 * kh;
      tms[row][nt * 16 + idx] = accv[r] + bias + xs[row][nt * 16 + idx];
    }
  }
  __syncthreads();

  // ---- LN2 -> write back to tokens ----
  if (tid < 16) {
    float mu = 0.f;
#pragma unroll
    for (int d = 0; d < 64; ++d) mu += tms[tid][d];
    mu *= (1.f / 64.f);
    float var = 0.f;
#pragma unroll
    for (int d = 0; d < 64; ++d) { const float t = tms[tid][d] - mu; var += t * t; }
    mu_s[tid] = mu;
    rs_s[tid] = rsqrtf(var * (1.f / 64.f) + 1e-5f);
  }
  __syncthreads();
  for (int i = tid; i < 1024; i += 128) {
    const int r = i >> 6, d = i & 63;
    xrow[i] = (tms[r][d] - mu_s[r]) * rs_s[r] * ln2g[d] + ln2b[d];
  }
}

// ---------------- Stage 8: classifier head ----------------
__global__ __launch_bounds__(256) void clf_kernel(
    const float* __restrict__ tokens, const float* __restrict__ w,
    const float* __restrict__ bb, float* __restrict__ out) {
  const int i = blockIdx.x * blockDim.x + threadIdx.x;
  if (i >= cB * 3) return;
  const int b = i / 3, c = i % 3;
  const float* x = tokens + (size_t)b * 16 * 64;   // token 0
  float acc = bb[c];
#pragma unroll
  for (int d = 0; d < 64; ++d) acc += x[d] * w[c * 64 + d];
  out[i] = acc;
}

// ---------------- host-side launcher ----------------
extern "C" void kernel_launch(void* const* d_in, const int* in_sizes, int n_in,
                              void* d_out, int out_size, void* d_ws, size_t ws_size,
                              hipStream_t stream) {
  (void)in_sizes; (void)n_in; (void)out_size; (void)ws_size;
  const float* X       = (const float*)d_in[0];
  const float* A_hat   = (const float*)d_in[1];
  const float* conv1_w = (const float*)d_in[2];
  const float* conv1_b = (const float*)d_in[3];
  const float* conv2_w = (const float*)d_in[4];
  const float* conv2_b = (const float*)d_in[5];
  const float* g1_W    = (const float*)d_in[6];
  const float* g1_q    = (const float*)d_in[7];
  const float* g1_bn_g = (const float*)d_in[8];
  const float* g1_bn_b = (const float*)d_in[9];
  const float* g1_bn_m = (const float*)d_in[10];
  const float* g1_bn_v = (const float*)d_in[11];
  const float* g2_W    = (const float*)d_in[12];
  const float* g2_q    = (const float*)d_in[13];
  const float* g2_bn_g = (const float*)d_in[14];
  const float* g2_bn_b = (const float*)d_in[15];
  const float* g2_bn_m = (const float*)d_in[16];
  const float* g2_bn_v = (const float*)d_in[17];
  const float* cls_tok = (const float*)d_in[18];
  const float* pos_emb = (const float*)d_in[19];
  const float* t_wqkv  = (const float*)d_in[20];
  const float* t_bqkv  = (const float*)d_in[21];
  const float* t_wo    = (const float*)d_in[22];
  const float* t_bo    = (const float*)d_in[23];
  const float* t_ln1g  = (const float*)d_in[24];
  const float* t_ln1b  = (const float*)d_in[25];
  const float* t_w1    = (const float*)d_in[26];
  const float* t_b1    = (const float*)d_in[27];
  const float* t_w2    = (const float*)d_in[28];
  const float* t_b2    = (const float*)d_in[29];
  const float* t_ln2g  = (const float*)d_in[30];
  const float* t_ln2b  = (const float*)d_in[31];
  const float* clf_w   = (const float*)d_in[32];
  const float* clf_b   = (const float*)d_in[33];

  char* ws = (char*)d_ws;
  size_t off = 0;
  auto wsalloc = [&](size_t bytes) -> char* {
    char* p = ws + off;
    off += (bytes + 255) & ~(size_t)255;
    return p;
  };
  // activations
  _Float16* feath = (_Float16*)wsalloc((size_t)cM * 64 * 2);
  float*    Y1    = (float*)   wsalloc((size_t)cM * 128 * 4);
  _Float16* H1h   = (_Float16*)wsalloc((size_t)cM * 128 * 2);
  float*    Y2    = (float*)   wsalloc((size_t)cM * 64 * 4);
  float*    H2    = (float*)   wsalloc((size_t)cM * 64 * 4);
  float*    toks  = (float*)   wsalloc((size_t)cB * 16 * 64 * 4);
  // f16 weight caches
  _Float16* w1h   = (_Float16*)wsalloc(32 * 32 * 2);
  _Float16* w2h   = (_Float16*)wsalloc(64 * 288 * 2);
  _Float16* g1h   = (_Float16*)wsalloc(128 * 64 * 2);
  _Float16* g2h   = (_Float16*)wsalloc(64 * 128 * 2);
  _Float16* qkvh  = (_Float16*)wsalloc((size_t)4 * 192 * 64 * 2);
  _Float16* woh   = (_Float16*)wsalloc((size_t)4 * 64 * 64 * 2);
  _Float16* w1th  = (_Float16*)wsalloc((size_t)4 * 128 * 64 * 2);
  _Float16* w2th  = (_Float16*)wsalloc((size_t)4 * 64 * 128 * 2);

  // 0) one-shot weight conversion (tiny)
  pad_w1_kernel<<<4, 256, 0, stream>>>(conv1_w, w1h);
  cvt_kernel<<<(64 * 288 + 255) / 256, 256, 0, stream>>>(conv2_w, w2h, 64 * 288);
  cvt_kernel<<<(128 * 64 + 255) / 256, 256, 0, stream>>>(g1_W, g1h, 128 * 64);
  cvt_kernel<<<(64 * 128 + 255) / 256, 256, 0, stream>>>(g2_W, g2h, 64 * 128);
  cvt_kernel<<<(4 * 192 * 64 + 255) / 256, 256, 0, stream>>>(t_wqkv, qkvh, 4 * 192 * 64);
  cvt_kernel<<<(4 * 64 * 64 + 255) / 256, 256, 0, stream>>>(t_wo, woh, 4 * 64 * 64);
  cvt_kernel<<<(4 * 128 * 64 + 255) / 256, 256, 0, stream>>>(t_w1, w1th, 4 * 128 * 64);
  cvt_kernel<<<(4 * 64 * 128 + 255) / 256, 256, 0, stream>>>(t_w2, w2th, 4 * 64 * 128);

  // 1) temporal CNN (WMMA implicit-GEMM convs)
  temporal_kernel<<<cM, 128, 0, stream>>>(X, w1h, conv1_b, w2h, conv2_b, feath);

  // 2) Y1 = feat @ g1_W^T   (M=18240, K=64, N=128)
  {
    const int tiles = (cM / 16) * (cHID / 16);
    gemm_f16w<<<(tiles + 3) / 4, 128, 0, stream>>>(feath, g1h, Y1, cM, cFIN, cHID, tiles);
  }
  // 3) H1 = relu(BN(Abar1 @ Y1))  (only f16 needed downstream)
  gcn_mix_kernel<<<cB * cSEG, 128, 0, stream>>>(A_hat, g1_q, Y1, g1_bn_g, g1_bn_b, g1_bn_m, g1_bn_v,
                                                nullptr, H1h, cHID);
  // 4) Y2 = H1 @ g2_W^T   (M=18240, K=128, N=64)
  {
    const int tiles = (cM / 16) * (cFIN / 16);
    gemm_f16w<<<(tiles + 3) / 4, 128, 0, stream>>>(H1h, g2h, Y2, cM, cHID, cFIN, tiles);
  }
  // 5) H2 = relu(BN(Abar2 @ Y2))  (only f32 needed downstream)
  gcn_mix_kernel<<<cB * cSEG, 128, 0, stream>>>(A_hat, g2_q, Y2, g2_bn_g, g2_bn_b, g2_bn_m, g2_bn_v,
                                                H2, nullptr, cFIN);
  // 6) tokens
  tokens_kernel<<<cB, 64, 0, stream>>>(H2, cls_tok, pos_emb, toks);

  // 7) transformer layers
  for (int l = 0; l < 4; ++l) {
    tx_layer<<<cB, 128, 0, stream>>>(toks,
        qkvh + (size_t)l * 192 * 64, t_bqkv + (size_t)l * 192,
        woh  + (size_t)l * 64 * 64,  t_bo   + (size_t)l * 64,
        t_ln1g + (size_t)l * 64,     t_ln1b + (size_t)l * 64,
        w1th + (size_t)l * 128 * 64, t_b1   + (size_t)l * 128,
        w2th + (size_t)l * 64 * 128, t_b2   + (size_t)l * 64,
        t_ln2g + (size_t)l * 64,     t_ln2b + (size_t)l * 64);
  }

  // 8) classifier
  clf_kernel<<<1, 256, 0, stream>>>(toks, clf_w, clf_b, (float*)d_out);
}
